// CausalSelfAttention_76227079570090
// MI455X (gfx1250) — matrix operand order repaired and checked by reading, not compiled, not run on hardware
//
#include <hip/hip_runtime.h>
#include <hip/hip_bf16.h>

// ---------------------------------------------------------------------------
// Causal GQA self-attention block for MI455X (gfx1250, wave32, WMMA bf16).
//   B=2, S=2048, H=2048, NH=16, NKV=4, HD=128
// Compute: bf16 inputs, f32 accumulation via v_wmma_f32_16x16x32_bf16.
// Data movement: async global->LDS (ASYNCcnt) for row-major tiles,
// transpose-on-store LDS staging for K-strided operands so every WMMA
// fragment read is a contiguous ds_load_b128.
// ---------------------------------------------------------------------------

#define B_    2
#define S_    2048
#define H_    2048
#define NH_   16
#define NKV_  4
#define HD_   128
#define DQ_   (NH_ * HD_)   // 2048
#define DKV_  (NKV_ * HD_)  // 512
#define EPSF  1e-6f

typedef __attribute__((ext_vector_type(16))) __bf16 v16bf;
typedef __attribute__((ext_vector_type(8)))  float  v8f;

union AF { v16bf v; uint4 q[2]; unsigned int u[8]; unsigned short h[16]; };
union CF { v8f   v; float f[8]; };

__device__ __forceinline__ unsigned short f2bf(float f) {
  unsigned int u = __float_as_uint(f);
  return (unsigned short)((u + 0x7FFFu + ((u >> 16) & 1u)) >> 16);  // RNE
}

// CDNA5 async global->LDS copy (16B per lane), tracked with ASYNCcnt.
__device__ __forceinline__ void async_b128_to_lds(const void* gptr, void* lptr) {
  unsigned           lds = (unsigned)(size_t)lptr;          // low 32b = LDS offset
  unsigned long long ga  = (unsigned long long)(size_t)gptr;
  asm volatile("global_load_async_to_lds_b128 %0, %1, off"
               :: "v"(lds), "v"(ga) : "memory");
}
__device__ __forceinline__ void wait_async0() {
  asm volatile("s_wait_asynccnt 0x0" ::: "memory");
}

// ---------------------------------------------------------------------------
// f32 -> bf16 converter (grid-stride)
// ---------------------------------------------------------------------------
__global__ void f32_to_bf16_kernel(const float* __restrict__ in,
                                   unsigned short* __restrict__ out, int n) {
  int i = blockIdx.x * blockDim.x + threadIdx.x;
  int stride = gridDim.x * blockDim.x;
  for (; i < n; i += stride) out[i] = f2bf(in[i]);
}

// ---------------------------------------------------------------------------
// Tiled WMMA GEMM: C(f32, MxN) = A(bf16, MxK, rm) * B(bf16, KxN, rm)
// Block tile 64x64, 8 waves (4x2), each wave owns 16x32 (2 accumulators).
// A tile staged via async global->LDS (row-major); B tile staged TRANSPOSED
// (Bst[n][k]) so B-fragment reads are two contiguous ds_load_b128 per frag.
// ---------------------------------------------------------------------------
__global__ __launch_bounds__(256) void gemm_bf16_wmma(
    const unsigned short* __restrict__ A, const unsigned short* __restrict__ Bm,
    float* __restrict__ C, int M, int N, int K) {
  __shared__ unsigned short As[64 * 32];    // 4 KB, [m][k]
  __shared__ unsigned short Bst[64 * 32];   // 4 KB, [n][k]  (transposed)

  const int tid  = threadIdx.x;
  const int lane = tid & 31;
  const int wave = tid >> 5;
  const int wm   = wave >> 1;
  const int wn   = wave & 1;
  const int bm   = blockIdx.y * 64;
  const int bn   = blockIdx.x * 64;

  // cooperative loader coordinates (8 ushorts = 16B per thread)
  const int ar = tid >> 2,  ac = (tid & 3) * 8;   // A tile 64x32
  const int br = tid >> 3,  bc = (tid & 7) * 8;   // B tile 32x64 (global layout)

  // WMMA per-lane fragment coordinates
  const int r    = lane & 15;
  const int half = lane >> 4;
  const int kbA  = half * 8;    // A-frag K sub-base
  const int kbB  = half * 8;    // B-frag K sub-base

  CF acc0, acc1;
#pragma unroll
  for (int j = 0; j < 8; ++j) { acc0.f[j] = 0.f; acc1.f[j] = 0.f; }

  for (int k0 = 0; k0 < K; k0 += 32) {
    // A tile: async DMA straight into LDS (no VGPR round trip)
    async_b128_to_lds(A + (size_t)(bm + ar) * K + k0 + ac, &As[ar * 32 + ac]);

    // B tile: coalesced global read, transpose-on-store into Bst[n][k]
    {
      uint4 bv = *(const uint4*)(Bm + (size_t)(k0 + br) * N + bn + bc);
      const unsigned short* bh = (const unsigned short*)&bv;
#pragma unroll
      for (int j = 0; j < 8; ++j) Bst[(bc + j) * 32 + br] = bh[j];
    }
    if (k0 + 32 < K) {
      __builtin_prefetch(A + (size_t)(bm + ar) * K + k0 + 32 + ac, 0, 3);
      __builtin_prefetch(Bm + (size_t)(k0 + 32 + br) * N + bn + bc, 0, 3);
    }
    wait_async0();
    __syncthreads();

    // A fragment: two contiguous b128 reads
    AF a;
    const unsigned short* arow = &As[(wm * 16 + r) * 32];
    a.q[0] = *(const uint4*)(arow + kbA);
    a.q[1] = *(const uint4*)(arow + 16 + kbA);

    // B fragments: two contiguous b128 reads each (thanks to transpose)
    AF b0, b1;
    const unsigned short* b0row = &Bst[(wn * 32 + r) * 32 + kbB];
    const unsigned short* b1row = &Bst[(wn * 32 + 16 + r) * 32 + kbB];
    b0.q[0] = *(const uint4*)(b0row);
    b0.q[1] = *(const uint4*)(b0row + 16);
    b1.q[0] = *(const uint4*)(b1row);
    b1.q[1] = *(const uint4*)(b1row + 16);

    acc0.v = __builtin_amdgcn_wmma_f32_16x16x32_bf16(false, a.v, false, b0.v,
                                                     (short)0, acc0.v, false, false);
    acc1.v = __builtin_amdgcn_wmma_f32_16x16x32_bf16(false, a.v, false, b1.v,
                                                     (short)0, acc1.v, false, false);
    __syncthreads();
  }

  // D layout: N = lane&15, M = j + 8*half
#pragma unroll
  for (int j = 0; j < 8; ++j) {
    size_t row = (size_t)(bm + wm * 16 + j + half * 8);
    int    col = bn + wn * 32 + r;
    C[row * N + col]      = acc0.f[j];
    C[row * N + col + 16] = acc1.f[j];
  }
}

// ---------------------------------------------------------------------------
// Fused per-head RMSNorm + RoPE, f32 in -> bf16 out.
// One block (128 threads) per (b, s, head) row of HD=128 elements.
// ---------------------------------------------------------------------------
__global__ __launch_bounds__(128) void rmsnorm_rope_kernel(
    const float* __restrict__ in, const float* __restrict__ w,
    const float* __restrict__ cosT, const float* __restrict__ sinT,
    unsigned short* __restrict__ out, int nheads) {
  __shared__ float row[HD_];
  __shared__ float red[4];
  const int rix = blockIdx.x;               // linear (b*S + s)*nheads + h
  const int d   = threadIdx.x;
  const int s   = (rix / nheads) % S_;

  float x = in[(size_t)rix * HD_ + d];
  row[d] = x;
  float ss = x * x;
#pragma unroll
  for (int m = 16; m >= 1; m >>= 1) ss += __shfl_xor(ss, m, 32);
  if ((d & 31) == 0) red[d >> 5] = ss;
  __syncthreads();

  float inv = rsqrtf((red[0] + red[1] + red[2] + red[3]) * (1.0f / HD_) + EPSF);
  float xn  = x * inv * w[d];
  float xo  = (d < 64) ? (-row[d + 64] * inv * w[d + 64])
                       : ( row[d - 64] * inv * w[d - 64]);
  float c  = cosT[s * HD_ + d];
  float sn = sinT[s * HD_ + d];
  out[(size_t)rix * HD_ + d] = f2bf(xn * c + xo * sn);
}

// ---------------------------------------------------------------------------
// Flash attention (causal, GQA). Block = (128 q rows) x (1 head) x (1 batch).
// 8 waves, each owns 16 q rows; Q fragments resident in VGPRs for whole loop.
// K tile (32x128, row-major) staged via async global->LDS; V tile staged
// TRANSPOSED (Vt[d][k], 128x32) so PV B-fragments are contiguous b128 reads.
// Online softmax; P routed D-layout -> A-layout via per-wave LDS scratch.
// 16 WMMA per 32-k step.
// ---------------------------------------------------------------------------
__global__ __launch_bounds__(256) void flash_attn_kernel(
    const unsigned short* __restrict__ Qb, const unsigned short* __restrict__ Kb,
    const unsigned short* __restrict__ Vb, unsigned short* __restrict__ Yb) {
  __shared__ unsigned short Ks[32 * HD_];        // 8 KB, [k][d]
  __shared__ unsigned short Vt[HD_ * 32];        // 8 KB, [d][k] (transposed)
  __shared__ unsigned short Ps[8][16 * 32];      // 8 KB (per-wave P scratch)

  const int tid  = threadIdx.x;
  const int lane = tid & 31;
  const int wave = tid >> 5;
  const int b    = blockIdx.z;
  const int h    = blockIdx.y;
  const int kvh  = h / (NH_ / NKV_);
  const int qb0  = blockIdx.x * 128;
  const int qrow = qb0 + wave * 16;

  const int r    = lane & 15;
  const int half = lane >> 4;
  const int kbA  = half * 8;
  const int kbB  = half * 8;

  // Q fragments: 4 chunks of K=32 covering HD=128 (resident for whole loop)
  AF aq[4];
  {
    const unsigned short* qp = Qb + ((size_t)(b * S_ + qrow + r) * NH_ + h) * HD_;
#pragma unroll
    for (int c = 0; c < 4; ++c) {
      aq[c].q[0] = *(const uint4*)(qp + 32 * c + kbA);
      aq[c].q[1] = *(const uint4*)(qp + 32 * c + 16 + kbA);
    }
  }

  CF o[8];
  float m_run[8], l_run[8];
#pragma unroll
  for (int cc = 0; cc < 8; ++cc)
#pragma unroll
    for (int j = 0; j < 8; ++j) o[cc].f[j] = 0.f;
#pragma unroll
  for (int j = 0; j < 8; ++j) { m_run[j] = -3.0e38f; l_run[j] = 0.f; }

  const int   kmax  = qb0 + 128;                 // causal block bound
  const float scale = 0.08838834764831845f;      // 1/sqrt(128)

  const int lr = tid >> 3;            // tile loader: k row 0..31
  const int lc = (tid & 7) * 16;      // d col 0..112

  for (int ks = 0; ks < kmax; ks += 32) {
    const size_t base = ((size_t)(b * S_ + ks + lr) * NKV_ + kvh) * HD_ + lc;

    // K tile: async DMA into LDS (row-major, consumed along d => contiguous)
    async_b128_to_lds(Kb + base,     &Ks[lr * HD_ + lc]);
    async_b128_to_lds(Kb + base + 8, &Ks[lr * HD_ + lc + 8]);

    // V tile: coalesced global read, transpose-on-store into Vt[d][k]
    {
      uint4 va = *(const uint4*)(Vb + base);
      uint4 vb = *(const uint4*)(Vb + base + 8);
      const unsigned short* ha = (const unsigned short*)&va;
      const unsigned short* hb = (const unsigned short*)&vb;
#pragma unroll
      for (int j = 0; j < 8; ++j) {
        Vt[(lc + j) * 32 + lr]     = ha[j];
        Vt[(lc + 8 + j) * 32 + lr] = hb[j];
      }
    }
    wait_async0();
    __syncthreads();

    // ---- scores: two 16x16 tiles (k 0..15 and 16..31), 8 WMMA ----
    CF s0, s1;
#pragma unroll
    for (int j = 0; j < 8; ++j) { s0.f[j] = 0.f; s1.f[j] = 0.f; }
#pragma unroll
    for (int c = 0; c < 4; ++c) {
      AF b0, b1;
      const unsigned short* k0p = &Ks[r        * HD_ + 32 * c + kbB];
      const unsigned short* k1p = &Ks[(16 + r) * HD_ + 32 * c + kbB];
      b0.q[0] = *(const uint4*)(k0p);
      b0.q[1] = *(const uint4*)(k0p + 16);
      b1.q[0] = *(const uint4*)(k1p);
      b1.q[1] = *(const uint4*)(k1p + 16);
      s0.v = __builtin_amdgcn_wmma_f32_16x16x32_bf16(false, aq[c].v, false, b0.v,
                                                     (short)0, s0.v, false, false);
      s1.v = __builtin_amdgcn_wmma_f32_16x16x32_bf16(false, aq[c].v, false, b1.v,
                                                     (short)0, s1.v, false, false);
    }

    // ---- causal mask + online softmax (row reductions via wave shuffles) ----
    float alpha[8];
    CF p0, p1;
#pragma unroll
    for (int j = 0; j < 8; ++j) {
      int   q_idx = qrow + j + half * 8;
      float v0 = s0.f[j] * scale;
      float v1 = s1.f[j] * scale;
      if (ks + r      > q_idx) v0 = -3.0e38f;
      if (ks + 16 + r > q_idx) v1 = -3.0e38f;
      float mx = fmaxf(v0, v1);
#pragma unroll
      for (int t = 8; t >= 1; t >>= 1) mx = fmaxf(mx, __shfl_xor(mx, t, 16));
      float mnew = fmaxf(m_run[j], mx);
      float a    = __expf(m_run[j] - mnew);
      float e0   = __expf(v0 - mnew);
      float e1   = __expf(v1 - mnew);
      float sum  = e0 + e1;
#pragma unroll
      for (int t = 8; t >= 1; t >>= 1) sum += __shfl_xor(sum, t, 16);
      l_run[j] = l_run[j] * a + sum;
      m_run[j] = mnew;
      alpha[j] = a;
      p0.f[j] = e0;
      p1.f[j] = e1;
    }

    // ---- P: D-layout -> A-layout through per-wave LDS scratch ----
    unsigned short* ps = Ps[wave];
#pragma unroll
    for (int j = 0; j < 8; ++j) {
      int m = j + half * 8;
      ps[m * 32 + r]      = f2bf(p0.f[j]);
      ps[m * 32 + 16 + r] = f2bf(p1.f[j]);
    }
    AF ap;
    {
      const unsigned short* prow = &ps[r * 32];
      ap.q[0] = *(const uint4*)(prow + kbA);
      ap.q[1] = *(const uint4*)(prow + 16 + kbA);
    }

    // ---- rescale O, then O += P @ V  (8 WMMA over the 8 d-chunks) ----
#pragma unroll
    for (int cc = 0; cc < 8; ++cc) {
#pragma unroll
      for (int j = 0; j < 8; ++j) o[cc].f[j] *= alpha[j];
      AF bv;
      const unsigned short* vrow = &Vt[(cc * 16 + r) * 32 + kbB];
      bv.q[0] = *(const uint4*)(vrow);
      bv.q[1] = *(const uint4*)(vrow + 16);
      o[cc].v = __builtin_amdgcn_wmma_f32_16x16x32_bf16(false, ap.v, false, bv.v,
                                                        (short)0, o[cc].v, false, false);
    }
    __syncthreads();
  }

  // ---- epilogue: 1/l scaling, bf16 store to (b,s,h,d) ----
#pragma unroll
  for (int j = 0; j < 8; ++j) {
    int   m    = j + half * 8;
    float invl = 1.0f / l_run[j];
    size_t base = ((size_t)(b * S_ + qrow + m) * NH_ + h) * HD_;
#pragma unroll
    for (int cc = 0; cc < 8; ++cc)
      Yb[base + cc * 16 + r] = f2bf(o[cc].f[j] * invl);
  }
}

// ---------------------------------------------------------------------------
// Host-side orchestration
// ---------------------------------------------------------------------------
extern "C" void kernel_launch(void* const* d_in, const int* in_sizes, int n_in,
                              void* d_out, int out_size, void* d_ws, size_t ws_size,
                              hipStream_t stream) {
  (void)in_sizes; (void)n_in; (void)out_size; (void)ws_size;
  const float* x  = (const float*)d_in[0];
  const float* rc = (const float*)d_in[1];
  const float* rs = (const float*)d_in[2];
  const float* Wq = (const float*)d_in[3];
  const float* Wk = (const float*)d_in[4];
  const float* Wv = (const float*)d_in[5];
  const float* Wo = (const float*)d_in[6];
  const float* qw = (const float*)d_in[7];
  const float* kw = (const float*)d_in[8];
  float* out = (float*)d_out;

  char*  ws  = (char*)d_ws;
  size_t off = 0;
  auto alloc = [&](size_t bytes) -> char* {
    char* p = ws + off;
    off += (bytes + 255) & ~(size_t)255;
    return p;
  };

  const int M = B_ * S_;  // 4096 token rows
  unsigned short* xb  = (unsigned short*)alloc((size_t)M * H_ * 2);
  unsigned short* Wqb = (unsigned short*)alloc((size_t)H_ * DQ_ * 2);
  unsigned short* Wkb = (unsigned short*)alloc((size_t)H_ * DKV_ * 2);
  unsigned short* Wvb = (unsigned short*)alloc((size_t)H_ * DKV_ * 2);
  unsigned short* Wob = (unsigned short*)alloc((size_t)H_ * H_ * 2);
  float*          Qf  = (float*)alloc((size_t)M * DQ_ * 4);
  float*          Kf  = (float*)alloc((size_t)M * DKV_ * 4);
  float*          Vf  = (float*)alloc((size_t)M * DKV_ * 4);
  unsigned short* Qh  = (unsigned short*)alloc((size_t)M * DQ_ * 2);
  unsigned short* Kh  = (unsigned short*)alloc((size_t)M * DKV_ * 2);
  unsigned short* Vh  = (unsigned short*)alloc((size_t)M * DKV_ * 2);
  unsigned short* Yb  = (unsigned short*)alloc((size_t)M * DQ_ * 2);

  auto conv = [&](const float* src, unsigned short* dst, int n) {
    int blocks = (n + 1023) / 1024;
    if (blocks > 4096) blocks = 4096;
    f32_to_bf16_kernel<<<blocks, 1024, 0, stream>>>(src, dst, n);
  };

  // 1) precision conversion of activations + weights
  conv(x,  xb,  M * H_);
  conv(Wq, Wqb, H_ * DQ_);
  conv(Wk, Wkb, H_ * DKV_);
  conv(Wv, Wvb, H_ * DKV_);
  conv(Wo, Wob, H_ * H_);

  // 2) QKV projections (WMMA bf16, f32 accumulate)
  gemm_bf16_wmma<<<dim3(DQ_ / 64,  M / 64), 256, 0, stream>>>(xb, Wqb, Qf, M, DQ_,  H_);
  gemm_bf16_wmma<<<dim3(DKV_ / 64, M / 64), 256, 0, stream>>>(xb, Wkb, Kf, M, DKV_, H_);
  gemm_bf16_wmma<<<dim3(DKV_ / 64, M / 64), 256, 0, stream>>>(xb, Wvb, Vf, M, DKV_, H_);

  // 3) fused RMSNorm + RoPE (Q, K) and bf16 conversion (V)
  rmsnorm_rope_kernel<<<M * NH_,  128, 0, stream>>>(Qf, qw, rc, rs, Qh, NH_);
  rmsnorm_rope_kernel<<<M * NKV_, 128, 0, stream>>>(Kf, kw, rc, rs, Kh, NKV_);
  conv(Vf, Vh, M * DKV_);

  // 4) causal GQA flash attention
  flash_attn_kernel<<<dim3(S_ / 128, NH_, B_), 256, 0, stream>>>(Qh, Kh, Vh, Yb);

  // 5) output projection -> f32 result
  gemm_bf16_wmma<<<dim3(H_ / 64, M / 64), 256, 0, stream>>>(Yb, Wob, out, M, H_, H_);
}
